// PatchMMD_SSIM_85435489452542
// MI455X (gfx1250) — compile-verified, matrix operand order checked by reading, not run on hardware
//
#include <hip/hip_runtime.h>
#include <hip/hip_bf16.h>

// ---------------------------------------------------------------------------
// PatchMMD-SSIM, fused for MI455X (gfx1250, wave32, WMMA).
//   G_c = Xs_c * Xs_c^T  (Xs = sqrt(win) * (x+1)/2 patch matrix, f16, K=128)
//   Symmetric: only lower-triangular tiles computed (2x off-diag weight).
//   SSIM map + s^T K s reduction fused into the WMMA epilogue.
// ---------------------------------------------------------------------------

typedef __attribute__((ext_vector_type(16))) _Float16 v16h;
typedef __attribute__((ext_vector_type(8)))  _Float16 v8h;
typedef __attribute__((ext_vector_type(8)))  float    v8f;

#define NPATCH   1225              // 35*35 patches per image
#define PREAL    2450              // 2*NPATCH combined rows
#define ROWS     2496              // padded to multiple of 64
#define KDIM     128               // 121 pixels padded to 128
#define C1F      (0.01f * 0.01f)
#define C2F      (0.03f * 0.03f)
#define WVAL     (1.0f / 1225.0f)

__device__ __forceinline__ float row_weight(int i) {
    // s vector: +1/N for first block, -1/M for second, 0 for padding rows.
    return (i < NPATCH) ? WVAL : ((i < PREAL) ? -WVAL : 0.0f);
}

// ---------------------------------------------------------------------------
// Kernel 1: patch extraction + per-patch stats.
// One thread per (batch, channel, row). Writes Xs row (f16, 128) and
// mu^2 / sigma (f32). Padding rows/cols are zero-filled.
// ---------------------------------------------------------------------------
__global__ void prep_kernel(const float* __restrict__ x,
                            const float* __restrict__ y,
                            _Float16* __restrict__ xs,
                            float* __restrict__ mu2,
                            float* __restrict__ sig) {
    int t = blockIdx.x * blockDim.x + threadIdx.x;
    if (t >= 2 * 3 * ROWS) return;
    int row = t % ROWS;
    int c   = (t / ROWS) % 3;
    int b   = t / (ROWS * 3);
    size_t base = (size_t)(b * 3 + c) * ROWS;
    _Float16* xrow = xs + (base + row) * KDIM;

    // 1D gaussian, sigma = 1.5, normalized.
    float g[11];
    float gs = 0.0f;
#pragma unroll
    for (int i = 0; i < 11; ++i) {
        float d = (float)i - 5.0f;
        g[i] = __expf(-(d * d) / 4.5f);
        gs += g[i];
    }
    float inv = 1.0f / gs;
#pragma unroll
    for (int i = 0; i < 11; ++i) g[i] *= inv;

    if (row < PREAL) {
        const float* img = (row < NPATCH) ? x : y;
        int p  = (row < NPATCH) ? row : row - NPATCH;
        int py = p / 35, px = p % 35;
        const float* ib = img + ((size_t)(b * 3 + c) * 80 + 2 * py) * 80 + 2 * px;
        float mu = 0.0f, s2 = 0.0f;
        for (int dy = 0; dy < 11; ++dy) {
#pragma unroll
            for (int dx = 0; dx < 11; ++dx) {
                float v = (ib[dy * 80 + dx] + 1.0f) * 0.5f;
                float w = g[dy] * g[dx];
                mu += w * v;
                s2 += w * v * v;
                xrow[dy * 11 + dx] = (_Float16)(v * __fsqrt_rn(w));
            }
        }
#pragma unroll
        for (int k = 121; k < KDIM; ++k) xrow[k] = (_Float16)0.0f;
        mu2[base + row] = mu * mu;
        sig[base + row] = s2 - mu * mu;
    } else {
        for (int k = 0; k < KDIM; ++k) xrow[k] = (_Float16)0.0f;
        mu2[base + row] = 0.0f;
        sig[base + row] = 0.0f;
    }
}

// ---------------------------------------------------------------------------
// Kernel 2: fused WMMA GEMM (3 channels) + SSIM map + weighted reduction.
// Grid: (ROWS/64, ROWS/32, B). 256 threads = 8 waves, each wave owns a
// 16x16 (i,j) tile; waves arranged 4x2 -> 64x32 per workgroup.
// Symmetry: tiles entirely above the diagonal (all i < j) are skipped and
// write a zero partial; per-element weight 2/1/0 for i>j / i==j / i<j.
// ---------------------------------------------------------------------------
__global__ __launch_bounds__(256) void gemm_ssim_kernel(
        const _Float16* __restrict__ xs,
        const float* __restrict__ mu2,
        const float* __restrict__ sig,
        float* __restrict__ partials) {
    const int b    = blockIdx.z;
    const int tid  = threadIdx.x;
    const size_t pIdx =
        ((size_t)b * gridDim.y + blockIdx.y) * gridDim.x + blockIdx.x;

    // Whole block above the diagonal -> nothing to do (covered by transpose).
    if ((int)blockIdx.y * 32 >= (int)blockIdx.x * 64 + 64) {
        if (tid == 0) partials[pIdx] = 0.0f;
        return;
    }

    const int wave = tid >> 5;
    const int lane = tid & 31;
    const int wr   = wave >> 1;          // 0..3
    const int wc   = wave & 1;           // 0..1
    const int iBase = blockIdx.x * 64 + wr * 16;
    const int jBase = blockIdx.y * 32 + wc * 16;
    const int lrow  = lane & 15;
    const int kh    = (lane >> 4) * 8;   // K half-offset per WMMA fragment layout

    v8f acc[3];
#pragma unroll
    for (int c = 0; c < 3; ++c) {
        const size_t cbase = (size_t)(b * 3 + c) * ROWS;
        const _Float16* Ap = xs + (cbase + iBase + lrow) * KDIM + kh;
        const _Float16* Bp = xs + (cbase + jBase + lrow) * KDIM + kh;
        // Preload all 4 K-steps' fragments (one big load clause), then issue
        // the 4 WMMAs back-to-back so loadcnt waits can be partial.
        v8h afr[8], bfr[8];
#pragma unroll
        for (int kk = 0; kk < 4; ++kk) {
            afr[2 * kk]     = *(const v8h*)(Ap + kk * 32);
            afr[2 * kk + 1] = *(const v8h*)(Ap + kk * 32 + 16);
            bfr[2 * kk]     = *(const v8h*)(Bp + kk * 32);
            bfr[2 * kk + 1] = *(const v8h*)(Bp + kk * 32 + 16);
        }
        v8f a_c = {};
#pragma unroll
        for (int kk = 0; kk < 4; ++kk) {
            v16h af, bf;
#pragma unroll
            for (int e = 0; e < 8; ++e) {
                af[e] = afr[2 * kk][e]; af[e + 8] = afr[2 * kk + 1][e];
                bf[e] = bfr[2 * kk][e]; bf[e + 8] = bfr[2 * kk + 1][e];
            }
            a_c = __builtin_amdgcn_wmma_f32_16x16x32_f16(
                false, af, false, bf, (short)0, a_c, false, false);
        }
        acc[c] = a_c;
    }

    // Epilogue: C/D layout -> element e of v8f in this lane is
    // (i = iBase + e + 8*(lane>=16), j = jBase + (lane&15)).
    const int jj   = jBase + lrow;
    const int moff = (lane >> 4) * 8;
    float part = 0.0f;
#pragma unroll
    for (int c = 0; c < 3; ++c) {
        const size_t cbase = (size_t)(b * 3 + c) * ROWS;
        const float m2j = mu2[cbase + jj];
        const float sgj = sig[cbase + jj];
        const float wj  = row_weight(jj);
#pragma unroll
        for (int e = 0; e < 8; ++e) {
            const int ii   = iBase + moff + e;
            const float m2i = mu2[cbase + ii];
            const float sgi = sig[cbase + ii];
            const float mm  = m2i * m2j;               // mus_mult
            const float sxy = acc[c][e] - mm;          // sigmas_xy
            const float ss  = ((2.0f * mm + C1F) / (m2i + m2j + C1F)) *
                              ((2.0f * sxy + C2F) / (sgi + sgj + C2F));
            const float symw = (ii > jj) ? 2.0f : ((ii == jj) ? 1.0f : 0.0f);
            part += symw * row_weight(ii) * wj * ss;
        }
    }
    part *= (1.0f / 3.0f);                             // channel mean

    // Wave reduce then workgroup reduce (deterministic; no atomics).
#pragma unroll
    for (int off = 16; off > 0; off >>= 1)
        part += __shfl_down(part, off, 32);
    __shared__ float wsum[8];
    if (lane == 0) wsum[wave] = part;
    __syncthreads();
    if (tid == 0) {
        float tot = 0.0f;
#pragma unroll
        for (int w = 0; w < 8; ++w) tot += wsum[w];
        partials[pIdx] = tot;
    }
}

// ---------------------------------------------------------------------------
// Kernel 3: deterministic final reduction, writes the scalar (batch mean).
// ---------------------------------------------------------------------------
__global__ __launch_bounds__(256) void final_reduce_kernel(
        const float* __restrict__ partials, int n, float* __restrict__ out) {
    __shared__ float sm[256];
    float t = 0.0f;
    for (int i = threadIdx.x; i < n; i += 256) t += partials[i];
    sm[threadIdx.x] = t;
    __syncthreads();
#pragma unroll
    for (int s = 128; s > 0; s >>= 1) {
        if (threadIdx.x < s) sm[threadIdx.x] += sm[threadIdx.x + s];
        __syncthreads();
    }
    if (threadIdx.x == 0) out[0] = sm[0] * 0.5f;   // mean over B=2
}

// ---------------------------------------------------------------------------
extern "C" void kernel_launch(void* const* d_in, const int* in_sizes, int n_in,
                              void* d_out, int out_size, void* d_ws, size_t ws_size,
                              hipStream_t stream) {
    const float* x = (const float*)d_in[0];
    const float* y = (const float*)d_in[1];
    float* out = (float*)d_out;

    // Workspace layout (all 256B-aligned regions):
    //   Xs   : 2*3*ROWS*KDIM f16   = 3,833,856 B
    //   mu2  : 2*3*ROWS f32        =    59,904 B
    //   sig  : 2*3*ROWS f32        =    59,904 B
    //   part : 39*78*2 f32         =    24,336 B
    char* ws = (char*)d_ws;
    const size_t XS_SZ  = (size_t)2 * 3 * ROWS * KDIM * sizeof(_Float16);
    const size_t ST_SZ  = (size_t)2 * 3 * ROWS * sizeof(float);
    _Float16* xs  = (_Float16*)(ws);
    float*    mu2 = (float*)(ws + XS_SZ);
    float*    sig = (float*)(ws + XS_SZ + ST_SZ);
    float*    prt = (float*)(ws + XS_SZ + 2 * ST_SZ);

    const int prepThreads = 2 * 3 * ROWS;
    prep_kernel<<<(prepThreads + 127) / 128, 128, 0, stream>>>(x, y, xs, mu2, sig);

    dim3 grid(ROWS / 64, ROWS / 32, 2);   // 39 x 78 x 2
    gemm_ssim_kernel<<<grid, 256, 0, stream>>>(xs, mu2, sig, prt);

    const int nPart = (ROWS / 64) * (ROWS / 32) * 2;  // 6084
    final_reduce_kernel<<<1, 256, 0, stream>>>(prt, nPart, out);
}
